// TraceDAE_22428319220502
// MI455X (gfx1250) — matrix operand order repaired and checked by reading, not compiled, not run on hardware
//
#include <hip/hip_runtime.h>

#define N_    8192
#define IND   128
#define HID_  64

typedef __attribute__((ext_vector_type(16))) __bf16 v16bf;
typedef __attribute__((ext_vector_type(8)))  float  v8f;

// ---- CDNA5 data-mover feature detection (compile-safe fallbacks) ----------
#if defined(__has_builtin)
# if __has_builtin(__builtin_amdgcn_tensor_load_to_lds)
#  define USE_TDM 1
# elif __has_builtin(__builtin_amdgcn_global_load_async_to_lds_b128)
#  define USE_ASYNC 1
# endif
#endif

__device__ __forceinline__ unsigned short f2bf(float f) {
  union { float f; unsigned u; } c; c.f = f;
  unsigned r = c.u + 0x7FFFu + ((c.u >> 16) & 1u);
  return (unsigned short)(r >> 16);
}
__device__ __forceinline__ float sigm(float x) { return 1.f / (1.f + __expf(-x)); }

// Low 32 bits of a generic pointer to LDS == wave-relative LDS byte address.
__device__ __forceinline__ unsigned lds_addr_of(const void* p) {
  return (unsigned)(uintptr_t)p;
}

#if defined(USE_TDM)
typedef unsigned __attribute__((ext_vector_type(4))) uvec4;
typedef int      __attribute__((ext_vector_type(4))) ivec4;
typedef int      __attribute__((ext_vector_type(8))) ivec8;

// One TDM descriptor: copy a 32(k) x 128(n) tile of 2-byte elements from
// WhT (row stride 8192 elements) into LDS, packed rows of 64B -> [n][k].
__device__ __forceinline__ void tdm_load_tile(unsigned ldsAddr,
                                              unsigned long long gAddr) {
  uvec4 g0;
  g0[0] = 1u;                                         // count=1, user mode
  g0[1] = ldsAddr;                                    // lds_addr
  g0[2] = (unsigned)(gAddr & 0xFFFFFFFFull);          // global_addr[31:0]
  g0[3] = (unsigned)((gAddr >> 32) & 0x01FFFFFFull)   // global_addr[56:32]
        | (2u << 30);                                 // type=2 ("image")
  ivec8 g1;
  g1[0] = (int)(1u << 16);       // data_size=1 (2 bytes), no multicast/flags
  g1[1] = (int)(8192u << 16);    // tensor_dim0 = 8192 (bits 63:48 = lo16)
  g1[2] = (int)(128u  << 16);    // dim0 hi16=0 | tensor_dim1=128 lo16
  g1[3] = (int)(32u   << 16);    // dim1 hi16=0 | tile_dim0=32
  g1[4] = 128;                   // tile_dim1=128, tile_dim2=0
  g1[5] = 8192;                  // tensor_dim0_stride lo32 = 8192
  g1[6] = 0;                     // stride hi16 | tensor_dim1_stride lo16
  g1[7] = 0;
  ivec4 z4 = {0, 0, 0, 0};
#if __clang_major__ >= 23
  ivec8 z8 = {0, 0, 0, 0, 0, 0, 0, 0};
  __builtin_amdgcn_tensor_load_to_lds(g0, g1, z4, z4, z8, 0);
#else
  __builtin_amdgcn_tensor_load_to_lds(g0, g1, z4, z4, 0);
#endif
}
#endif  // USE_TDM

#if defined(USE_ASYNC)
typedef __attribute__((address_space(3))) unsigned short       lds_us;
typedef __attribute__((address_space(1))) const unsigned short gbl_us;
__device__ __forceinline__ void async_stage(unsigned short* dst,
                                            const unsigned short* WhT,
                                            int j, int t) {
  // thread t copies row n=t: 4 x 16B chunks (32 bf16 of k) into [n][k] tile
#pragma unroll
  for (int c = 0; c < 4; ++c) {
    const int ko = c * 8;
    __builtin_amdgcn_global_load_async_to_lds_b128(
        (gbl_us*)&WhT[(size_t)t * N_ + j + ko],
        (lds_us*)&dst[t * 32 + ko], 0, 0);
  }
}
__device__ __forceinline__ void wait_async0() {
#if __has_builtin(__builtin_amdgcn_s_wait_asynccnt)
  __builtin_amdgcn_s_wait_asynccnt(0);
#else
  asm volatile("s_wait_asynccnt 0" ::: "memory");
#endif
}
#endif  // USE_ASYNC

// ---------------------------------------------------------------------------
// Kernel 1: per-row MLP + GAT projections + encoder/decoder LSTM (h0=c0=0).
// Writes Wh TRANSPOSED (col-major, [IND][N_]) in bf16 for the TDM/frag path.
// ---------------------------------------------------------------------------
__global__ __launch_bounds__(128)
void k_rows(const float* __restrict__ x,
            const float* __restrict__ feaW, const float* __restrict__ feaB,
            const float* __restrict__ gatW, const float* __restrict__ gatA,
            const float* __restrict__ encWih, const float* __restrict__ encBih,
            const float* __restrict__ encBhh,
            const float* __restrict__ decWih, const float* __restrict__ decBih,
            const float* __restrict__ decBhh,
            unsigned short* __restrict__ WhT,
            float* __restrict__ s1, float* __restrict__ s2,
            float* __restrict__ attOut) {
  const int row = blockIdx.x, t = threadIdx.x;
  __shared__ float xr[IND];
  __shared__ float st[HID_];
  __shared__ float red1[128], red2[128];
  __shared__ float gbuf[256];
  __shared__ float henc[HID_];
  __shared__ float dbuf[512];

  xr[t] = x[(size_t)row * IND + t];
  __syncthreads();

  if (t < HID_) {                       // structure = relu(x @ fea_W + fea_b)
    float a = feaB[t];
    for (int k = 0; k < IND; ++k) a += xr[k] * feaW[k * HID_ + t];
    st[t] = a > 0.f ? a : 0.f;
  }
  __syncthreads();

  float w = 0.f;                        // Wh = structure @ gat_W
  for (int h = 0; h < HID_; ++h) w += st[h] * gatW[h * IND + t];
  WhT[(size_t)t * N_ + row] = f2bf(w);  // column-major for TDM tile copies
  red1[t] = w * gatA[t];
  red2[t] = w * gatA[IND + t];
  __syncthreads();
  for (int s = 64; s > 0; s >>= 1) {
    if (t < s) { red1[t] += red1[t + s]; red2[t] += red2[t + s]; }
    __syncthreads();
  }
  if (t == 0) { s1[row] = red1[0]; s2[row] = red2[0]; }

  for (int gi = 0; gi < 2; ++gi) {      // encoder gates (h0 = c0 = 0)
    int g = t + gi * 128;
    float a = encBih[g] + encBhh[g];
    for (int k = 0; k < IND; ++k) a += xr[k] * encWih[g * IND + k];
    gbuf[g] = a;
  }
  __syncthreads();
  if (t < HID_) {
    float ig = sigm(gbuf[t]);
    float gg = tanhf(gbuf[2 * HID_ + t]);
    float og = sigm(gbuf[3 * HID_ + t]);
    henc[t] = og * tanhf(ig * gg);
  }
  __syncthreads();

  for (int gi = 0; gi < 4; ++gi) {      // decoder gates
    int g = t + gi * 128;
    float a = decBih[g] + decBhh[g];
    for (int h = 0; h < HID_; ++h) a += henc[h] * decWih[g * HID_ + h];
    dbuf[g] = a;
  }
  __syncthreads();
  {
    float i2 = sigm(dbuf[t]);
    float g2 = tanhf(dbuf[2 * IND + t]);
    float o2 = sigm(dbuf[3 * IND + t]);
    attOut[(size_t)row * IND + t] = o2 * tanhf(i2 * g2);
  }
}

// ---------------------------------------------------------------------------
// Kernel 2: fused GAT attention, single pass over adj, online softmax,
// P@Wh accumulated with v_wmma_f32_16x16x32_bf16. Wh k-tiles are staged into
// LDS with the Tensor Data Mover (double-buffered, s_wait_tensorcnt), falling
// back to async-LDS copies or VALU staging if the builtins are absent.
// ---------------------------------------------------------------------------
__global__ __launch_bounds__(128)
void k_attn(const float* __restrict__ adj,
            const unsigned short* __restrict__ WhT,
            const float* __restrict__ s1, const float* __restrict__ s2,
            unsigned short* __restrict__ embBf) {
  __shared__ __align__(16) unsigned short tileT[2][IND * 32];  // [n][k]

  const int t    = threadIdx.x;
  const int lane = t & 31;
  const int wv   = t >> 5;        // wave 0..3
  const int hi   = lane >> 4;     // half-wave
  const int lm   = lane & 15;     // M row / N col within 16-tile
  const int r0   = blockIdx.x * 64 + wv * 16;
  const int kb   = 8 * hi;        // A-frag K base (ISA 16-bit A layout)
  const int NSTEP = N_ / 32;

  const float s1v = s1[r0 + lm];
  float m = -INFINITY, l = 0.f;

  v8f acc[8];
  for (int ti = 0; ti < 8; ++ti)
    for (int r = 0; r < 8; ++r) acc[ti][r] = 0.f;

  const size_t arow = (size_t)(r0 + lm) * N_;

  // prologue: stage tile for j=0 into buffer 0
#if defined(USE_TDM)
  if (wv == 0)
    tdm_load_tile(lds_addr_of(&tileT[0][0]),
                  (unsigned long long)(uintptr_t)WhT);
#elif defined(USE_ASYNC)
  async_stage(&tileT[0][0], WhT, 0, t);
#endif

  for (int s = 0; s < NSTEP; ++s) {
    const int j   = s * 32;
    const int cur = s & 1;

#if defined(USE_TDM)
    __builtin_amdgcn_s_wait_tensorcnt(0);   // wave0 waits its DMA; others NOP
    __syncthreads();                         // tile[cur] visible to all waves
    if (wv == 0 && s + 1 < NSTEP)            // overlap next tile with compute
      tdm_load_tile(lds_addr_of(&tileT[cur ^ 1][0]),
                    (unsigned long long)(uintptr_t)WhT +
                        (unsigned long long)(j + 32) * 2ull);
#elif defined(USE_ASYNC)
    wait_async0();
    __syncthreads();
    if (s + 1 < NSTEP) async_stage(&tileT[cur ^ 1][0], WhT, j + 32, t);
#else
    __syncthreads();
    for (int idx = t; idx < IND * 32; idx += 128) {
      int n = idx >> 5, k = idx & 31;        // coalesced along k
      tileT[cur][n * 32 + k] = WhT[(size_t)n * N_ + j + k];
    }
    __syncthreads();
#endif
    const unsigned short* tb = tileT[cur];

    // keep the HBM adj stream deep: prefetch 8 j-steps (1KB/row) ahead
    if (j + 256 < N_) __builtin_prefetch(&adj[arow + j + 256 + kb], 0, 1);

    // scores for this wave's 16x32 tile, built in A-fragment order
    float e[16];
    float cmax = -INFINITY;
#pragma unroll
    for (int half = 0; half < 2; ++half) {
      const int ks = half * 16 + kb;
      float4 a0 = *(const float4*)&adj[arow + j + ks];
      float4 a1 = *(const float4*)&adj[arow + j + ks + 4];
      float4 z0 = *(const float4*)&s2[j + ks];
      float4 z1 = *(const float4*)&s2[j + ks + 4];
      float av[8] = {a0.x, a0.y, a0.z, a0.w, a1.x, a1.y, a1.z, a1.w};
      float zv[8] = {z0.x, z0.y, z0.z, z0.w, z1.x, z1.y, z1.z, z1.w};
#pragma unroll
      for (int q = 0; q < 8; ++q) {
        float v = s1v + zv[q];
        v = v > 0.f ? v : 0.2f * v;            // leaky_relu
        v = av[q] > 0.f ? v : -9.0e15f;        // mask (matches reference)
        e[half * 8 + q] = v;
        cmax = fmaxf(cmax, v);
      }
    }
    cmax = fmaxf(cmax, __shfl_xor(cmax, 16, 32));
    const float mnew = fmaxf(m, cmax);

    float psum = 0.f;
    union { v16bf v; unsigned short u[16]; } A;
#pragma unroll
    for (int q = 0; q < 16; ++q) {
      float p = __expf(e[q] - mnew);
      psum += p;
      A.u[q] = f2bf(p);
    }
    psum += __shfl_xor(psum, 16, 32);
    const float scale = __expf(m - mnew);
    l = l * scale + psum;
    m = mnew;

    // rescale C fragments: VGPR r of acc holds row (r + 8*hi)
#pragma unroll
    for (int r = 0; r < 8; ++r) {
      float sc = __shfl(scale, r + 8 * hi, 32);
#pragma unroll
      for (int ti = 0; ti < 8; ++ti) acc[ti][r] *= sc;
    }

    // 8 column tiles x (16x16x32) bf16 WMMA
#pragma unroll
    for (int ti = 0; ti < 8; ++ti) {
      union { v16bf v; uint4 q[2]; } B;
      const int n = ti * 16 + lm;
      B.q[0] = *(const uint4*)&tb[n * 32 + 8 * hi];
      B.q[1] = *(const uint4*)&tb[n * 32 + 16 + 8 * hi];
      acc[ti] = __builtin_amdgcn_wmma_f32_16x16x32_bf16(
          false, A.v, false, B.v, (short)0, acc[ti], false, false);
    }
  }

  // epilogue: normalize, elu, store emb as bf16
#pragma unroll
  for (int r = 0; r < 8; ++r) {
    const int row = r0 + r + 8 * hi;
    float lr = __shfl(l, r + 8 * hi, 32);
    float inv = (lr != 0.f) ? 1.f / lr : 0.f;
#pragma unroll
    for (int ti = 0; ti < 8; ++ti) {
      float v = acc[ti][r] * inv;
      v = v > 0.f ? v : (__expf(v) - 1.f);     // elu
      embBf[(size_t)row * IND + ti * 16 + lm] = f2bf(v);
    }
  }
}

// ---------------------------------------------------------------------------
// Kernel 3: con_adj = emb @ emb^T (write-bound: 256 MB out).
// 128x128 tile per 256-thread block; fragments load straight from row-major
// bf16 emb (Gram => B^T is row-major), K=128 in 4 WMMA steps.
// ---------------------------------------------------------------------------
__global__ __launch_bounds__(256)
void k_gram(const unsigned short* __restrict__ embBf, float* __restrict__ out) {
  const int t    = threadIdx.x;
  const int lane = t & 31;
  const int wv   = t >> 5;        // 0..7
  const int hi   = lane >> 4;
  const int lm   = lane & 15;
  const int R0   = blockIdx.y * 128 + wv * 16;
  const int C0   = blockIdx.x * 128;
  const int kb   = 8 * hi;

  v8f acc[8];
  for (int ti = 0; ti < 8; ++ti)
    for (int r = 0; r < 8; ++r) acc[ti][r] = 0.f;

  const size_t arow = (size_t)(R0 + lm) * IND;

  for (int ks = 0; ks < IND; ks += 32) {
    union { v16bf v; uint4 q[2]; } A;
    A.q[0] = *(const uint4*)&embBf[arow + ks + kb];
    A.q[1] = *(const uint4*)&embBf[arow + ks + 16 + kb];
#pragma unroll
    for (int ti = 0; ti < 8; ++ti) {
      union { v16bf v; uint4 q[2]; } B;
      const size_t brow = (size_t)(C0 + ti * 16 + lm) * IND;
      B.q[0] = *(const uint4*)&embBf[brow + ks + 8 * hi];
      B.q[1] = *(const uint4*)&embBf[brow + ks + 16 + 8 * hi];
      acc[ti] = __builtin_amdgcn_wmma_f32_16x16x32_bf16(
          false, A.v, false, B.v, (short)0, acc[ti], false, false);
    }
  }

#pragma unroll
  for (int r = 0; r < 8; ++r) {
    const size_t row = (size_t)(R0 + r + 8 * hi);
#pragma unroll
    for (int ti = 0; ti < 8; ++ti) {
      out[row * N_ + C0 + ti * 16 + lm] = acc[ti][r];
    }
  }
}

// ---------------------------------------------------------------------------
extern "C" void kernel_launch(void* const* d_in, const int* in_sizes, int n_in,
                              void* d_out, int out_size, void* d_ws, size_t ws_size,
                              hipStream_t stream) {
  (void)in_sizes; (void)n_in; (void)out_size; (void)ws_size;
  const float* x      = (const float*)d_in[0];
  const float* adj    = (const float*)d_in[1];
  const float* feaW   = (const float*)d_in[2];
  const float* feaB   = (const float*)d_in[3];
  const float* gatW   = (const float*)d_in[4];
  const float* gatA   = (const float*)d_in[5];
  const float* encWih = (const float*)d_in[6];   // d_in[7] enc_W_hh unused (h0=0)
  const float* encBih = (const float*)d_in[8];
  const float* encBhh = (const float*)d_in[9];
  const float* decWih = (const float*)d_in[10];  // d_in[11] dec_W_hh unused (h0=0)
  const float* decBih = (const float*)d_in[12];
  const float* decBhh = (const float*)d_in[13];

  char* ws = (char*)d_ws;
  unsigned short* WhT   = (unsigned short*)ws;  ws += (size_t)N_ * IND * sizeof(unsigned short);
  unsigned short* embBf = (unsigned short*)ws;  ws += (size_t)N_ * IND * sizeof(unsigned short);
  float* s1 = (float*)ws;                       ws += (size_t)N_ * sizeof(float);
  float* s2 = (float*)ws;                       ws += (size_t)N_ * sizeof(float);

  float* conOut = (float*)d_out;                // [N, N]
  float* attOut = conOut + (size_t)N_ * N_;     // [N, IN_DIM]

  k_rows<<<N_, 128, 0, stream>>>(x, feaW, feaB, gatW, gatA,
                                 encWih, encBih, encBhh,
                                 decWih, decBih, decBhh,
                                 WhT, s1, s2, attOut);
  k_attn<<<N_ / 64, 128, 0, stream>>>(adj, WhT, s1, s2, embBf);
  k_gram<<<dim3(64, 64), 256, 0, stream>>>(embBf, conOut);
}
